// DualViewAttention_7687991460298
// MI455X (gfx1250) — compile-verified, matrix operand order checked by reading, not run on hardware
//
#include <hip/hip_runtime.h>
#include <hip/hip_bf16.h>
#include <math.h>

typedef __attribute__((ext_vector_type(16))) __bf16 v16bf;
typedef __attribute__((ext_vector_type(8)))  __bf16 v8bf;
typedef __attribute__((ext_vector_type(8)))  float  v8f;

#define B_      65536
#define D_      512
#define H_      128
#define KSTEPS1 65          // (4*512 + 32) / 32
#define FSTRIDE 264         // 256 feats + 8 pad (halves) -> conflict-free b128 LDS loads
#define H1STRIDE 136        // 128 + 8 pad (halves)
#define H2STRIDE 132        // 128 + 4 pad (floats)
#define P1ELEMS (65*8*32*16)
#define P2ELEMS (4*8*32*16)

// ---------------- kernel 0: zero column-sum accumulators ----------------
__global__ void k_init(float* cs_a, float* cs_ae) {
  int t = blockIdx.x * 256 + threadIdx.x;
  if (t < D_) { cs_a[t] = 0.f; cs_ae[t] = 0.f; }
}

// ---------------- kernel 1: per-row stats + column sums of normalized rows ----------------
__global__ __launch_bounds__(256)
void k_stats(const float* __restrict__ A, const float* __restrict__ AE,
             float* __restrict__ rstats, float* __restrict__ cs_a, float* __restrict__ cs_ae) {
  __shared__ float lca[D_];
  __shared__ float lce[D_];
  int t = threadIdx.x;
  for (int q = 0; q < 2; ++q) { lca[t + 256*q] = 0.f; lce[t + 256*q] = 0.f; }
  __syncthreads();

  int wave = t >> 5, lane = t & 31;
  int rowbase = blockIdx.x * 64 + wave * 8;
  for (int rr = 0; rr < 8; ++rr) {
    int row = rowbase + rr;
    const float* pa = A  + (size_t)row * D_;
    const float* pe = AE + (size_t)row * D_;
    float va[16], ve[16];
    float ssa = 0.f, sse = 0.f, dt = 0.f;
#pragma unroll
    for (int i = 0; i < 16; ++i) {
      float a = pa[i*32 + lane], e = pe[i*32 + lane];
      va[i] = a; ve[i] = e;
      ssa += a*a; sse += e*e; dt += a*e;
    }
#pragma unroll
    for (int m = 16; m >= 1; m >>= 1) {
      ssa += __shfl_xor(ssa, m, 32);
      sse += __shfl_xor(sse, m, 32);
      dt  += __shfl_xor(dt,  m, 32);
    }
    if (lane == 0) {
      float* rp = rstats + (size_t)row * 4;
      rp[0] = ssa; rp[1] = sse; rp[2] = dt;
    }
    float inva = 1.f / fmaxf(sqrtf(ssa), 1e-12f);
    float inve = 1.f / fmaxf(sqrtf(sse), 1e-12f);
#pragma unroll
    for (int i = 0; i < 16; ++i) {
      atomicAdd(&lca[i*32 + lane], va[i] * inva);
      atomicAdd(&lce[i*32 + lane], ve[i] * inve);
    }
  }
  __syncthreads();
  for (int q = 0; q < 2; ++q) {
    int j = t + 256*q;
    atomicAdd(&cs_a[j],  lca[j]);
    atomicAdd(&cs_ae[j], lce[j]);
  }
}

// ---------------- kernel 2: normalize batch centers ----------------
__global__ void k_centers(const float* cs_a, const float* cs_ae, float* c_a, float* c_ae) {
  __shared__ float red[D_];
  int t = threadIdx.x;
  float ma = cs_a[t] * (1.f / (float)B_);
  red[t] = ma * ma; __syncthreads();
  for (int s = 256; s >= 1; s >>= 1) { if (t < s) red[t] += red[t + s]; __syncthreads(); }
  float na = sqrtf(red[0]);
  c_a[t] = ma / fmaxf(na, 1e-12f);
  __syncthreads();
  float me = cs_ae[t] * (1.f / (float)B_);
  red[t] = me * me; __syncthreads();
  for (int s = 256; s >= 1; s >>= 1) { if (t < s) red[t] += red[t + s]; __syncthreads(); }
  float ne = sqrtf(red[0]);
  c_ae[t] = me / fmaxf(ne, 1e-12f);
}

// ---------------- kernel 3: fold W1 -> combined/permuted bf16 B-operand layout; pack W2 ----------------
// packed index = ((kstep*8 + ntile)*32 + lane)*16 + half
// B-frag lane mapping: n = ntile*16 + (lane&15);  k32 = (half&7) + (half&8 ? 16 : 0) + (lane&16 ? 8 : 0)
// feature index f (k): f<2048: g=f>>8, r=f&255, type=r>>6, dd=r&63, d=g*64+dd
__global__ __launch_bounds__(256)
void k_pack(const float* __restrict__ W1, const float* __restrict__ W2,
            __bf16* __restrict__ p1, __bf16* __restrict__ p2) {
  int idx = blockIdx.x * 256 + threadIdx.x;
  if (idx < P1ELEMS) {
    int h = idx & 15, lane = (idx >> 4) & 31, nt = (idx >> 9) & 7, ks = idx >> 12;
    int n = nt * 16 + (lane & 15);
    int k32 = (h & 7) + ((h & 8) ? 16 : 0) + ((lane & 16) ? 8 : 0);
    int f = ks * 32 + k32;
    float v;
    if (f < 2048) {
      int g = f >> 8, r = f & 255, ty = r >> 6, dd = r & 63, d = g * 64 + dd;
      if (ty == 0)      v = W1[(size_t)d*H_ + n] + W1[(size_t)(1024+d)*H_ + n] + 0.5f*W1[(size_t)(2560+d)*H_ + n];
      else if (ty == 1) v = W1[(size_t)(512+d)*H_ + n] - W1[(size_t)(1024+d)*H_ + n] + 0.5f*W1[(size_t)(2560+d)*H_ + n];
      else if (ty == 2) v = W1[(size_t)(1536+d)*H_ + n];
      else              v = W1[(size_t)(2048+d)*H_ + n];
    } else {
      int j = f - 2048;
      v = (j < 10) ? W1[(size_t)(3072+j)*H_ + n] : 0.f;
    }
    p1[idx] = (__bf16)v;
  } else if (idx < P1ELEMS + P2ELEMS) {
    int i2 = idx - P1ELEMS;
    int h = i2 & 15, lane = (i2 >> 4) & 31, nt = (i2 >> 9) & 7, ks = i2 >> 12;
    int n = nt * 16 + (lane & 15);
    int k32 = (h & 7) + ((h & 8) ? 16 : 0) + ((lane & 16) ? 8 : 0);
    p2[i2] = (__bf16)W2[(size_t)(ks*32 + k32)*H_ + n];
  }
}

// ---------------- kernel 4: fused features + WMMA MLP + softmax ----------------
__global__ __launch_bounds__(256)
void k_main(const float* __restrict__ A, const float* __restrict__ AE,
            const float* __restrict__ b1, const float* __restrict__ b2,
            const float* __restrict__ W3, const float* __restrict__ b3,
            const float* __restrict__ rstats,
            const float* __restrict__ c_a, const float* __restrict__ c_ae,
            const __bf16* __restrict__ p1, const __bf16* __restrict__ p2,
            float* __restrict__ out) {
  __shared__ __attribute__((aligned(16))) __bf16 featLDS[64 * FSTRIDE];
  __shared__ __attribute__((aligned(16))) __bf16 h1LDS[64 * H1STRIDE];
  __shared__ float dotsLDS[64][4];
  __shared__ float rsLDS[64][4];

  int t = threadIdx.x;
  int wave = t >> 5, lane = t & 31;
  int row0 = blockIdx.x * 64;

  if (t < 64) {
    const float* rp = rstats + (size_t)(row0 + t) * 4;
    rsLDS[t][0] = rp[0]; rsLDS[t][1] = rp[1]; rsLDS[t][2] = rp[2];
  }
  __syncthreads();

  v8f acc[4];
#pragma unroll
  for (int s = 0; s < 4; ++s)
#pragma unroll
    for (int v = 0; v < 8; ++v) acc[s][v] = 0.f;

  // feature-build mapping: 4 threads per row
  int fr = t >> 2;            // row in tile
  int fq = t & 3;             // 16-d quarter of the 64-d chunk
  float inva, inve;
  {
    float ssa = rsLDS[fr][0], sse = rsLDS[fr][1];
    inva = 1.f / fmaxf(sqrtf(ssa), 1e-12f);
    inve = 1.f / fmaxf(sqrtf(sse), 1e-12f);
  }
  const float* pa = A  + (size_t)(row0 + fr) * D_ + fq * 16;
  const float* pe = AE + (size_t)(row0 + fr) * D_ + fq * 16;
  float pda = 0.f, pde = 0.f, pxa = 0.f, pxe = 0.f;   // center-dot partials

  for (int c = 0; c < 8; ++c) {
    __syncthreads();                       // featLDS free (prev chunk consumed)
    __bf16* fl = featLDS + fr * FSTRIDE;
#pragma unroll
    for (int i = 0; i < 16; ++i) {
      int dd = fq * 16 + i;
      int d  = c * 64 + dd;
      float ra = pa[c * 64 + i];
      float re = pe[c * 64 + i];
      float a = ra * inva, e = re * inve;
      fl[dd]        = (__bf16)a;
      fl[64 + dd]   = (__bf16)e;
      fl[128 + dd]  = (__bf16)fabsf(a - e);
      fl[192 + dd]  = (__bf16)(a * e);
      float ca = c_a[d], ce = c_ae[d];
      pda += ra * ca; pde += re * ce; pxa += ra * ce; pxe += re * ca;
    }
    __syncthreads();                       // features ready
    for (int ks = 0; ks < 8; ++ks) {
      int kg = c * 8 + ks;
      v16bf bfrag = *(const v16bf*)(p1 + ((size_t)(kg * 8 + wave) * 32 + lane) * 16);
      int koff = ks * 32 + ((lane & 16) ? 8 : 0);
#pragma unroll
      for (int s = 0; s < 4; ++s) {
        const __bf16* ap = featLDS + (s * 16 + (lane & 15)) * FSTRIDE + koff;
        v8bf lo = *(const v8bf*)ap;
        v8bf hi = *(const v8bf*)(ap + 16);
        v16bf af = __builtin_shufflevector(lo, hi, 0,1,2,3,4,5,6,7,8,9,10,11,12,13,14,15);
        acc[s] = __builtin_amdgcn_wmma_f32_16x16x32_bf16(false, af, false, bfrag,
                                                         (short)0, acc[s], false, false);
      }
    }
  }

  // reduce center-dot partials (4 consecutive lanes per row)
  pda += __shfl_xor(pda, 1, 32); pda += __shfl_xor(pda, 2, 32);
  pde += __shfl_xor(pde, 1, 32); pde += __shfl_xor(pde, 2, 32);
  pxa += __shfl_xor(pxa, 1, 32); pxa += __shfl_xor(pxa, 2, 32);
  pxe += __shfl_xor(pxe, 1, 32); pxe += __shfl_xor(pxe, 2, 32);
  __syncthreads();                         // last chunk's WMMA reads done
  if (fq == 0) { dotsLDS[fr][0] = pda; dotsLDS[fr][1] = pde; dotsLDS[fr][2] = pxa; dotsLDS[fr][3] = pxe; }
  __syncthreads();

  // reliability features -> featLDS[:, 0..31] (22 zero pad)
  if (t < 64) {
    float ssa = rsLDS[t][0], sse = rsLDS[t][1], dt = rsLDS[t][2];
    float na = sqrtf(ssa), ne = sqrtf(sse);
    float ia = 1.f / fmaxf(na, 1e-12f), ie = 1.f / fmaxf(ne, 1e-12f);
    float agree = dt * ia * ie;
    float esum = fmaxf(na + ne, 1e-6f);
    float egap = (na - ne) / esum;
    float nres = sqrtf(fmaxf(ssa + sse - 2.f * dt, 0.f)) / esum;
    float cca = dotsLDS[t][0] * ia, cce = dotsLDS[t][1] * ie;
    float xca = dotsLDS[t][2] * ia, xce = dotsLDS[t][3] * ie;
    float mga = cca - xca, mge = cce - xce;
    float rv[10] = { agree, egap, nres, fabsf(egap), na/esum, ne/esum, mga, mge, mga - mge, cca - cce };
    __bf16* fl = featLDS + t * FSTRIDE;
#pragma unroll
    for (int j = 0; j < 10; ++j) fl[j] = (__bf16)rv[j];
#pragma unroll
    for (int j = 10; j < 32; ++j) fl[j] = (__bf16)0.f;
  }
  __syncthreads();
  {
    int kg = KSTEPS1 - 1;                  // reliability K-step
    v16bf bfrag = *(const v16bf*)(p1 + ((size_t)(kg * 8 + wave) * 32 + lane) * 16);
    int koff = (lane & 16) ? 8 : 0;
#pragma unroll
    for (int s = 0; s < 4; ++s) {
      const __bf16* ap = featLDS + (s * 16 + (lane & 15)) * FSTRIDE + koff;
      v8bf lo = *(const v8bf*)ap;
      v8bf hi = *(const v8bf*)(ap + 16);
      v16bf af = __builtin_shufflevector(lo, hi, 0,1,2,3,4,5,6,7,8,9,10,11,12,13,14,15);
      acc[s] = __builtin_amdgcn_wmma_f32_16x16x32_bf16(false, af, false, bfrag,
                                                       (short)0, acc[s], false, false);
    }
  }

  // layer-1 epilogue: bias + leaky -> h1 (bf16)
  {
    float bn = b1[wave * 16 + (lane & 15)];
#pragma unroll
    for (int s = 0; s < 4; ++s)
#pragma unroll
      for (int v = 0; v < 8; ++v) {
        int rl = s * 16 + ((lane & 16) ? 8 : 0) + v;
        float x = acc[s][v] + bn;
        x = (x > 0.f) ? x : 0.2f * x;
        h1LDS[rl * H1STRIDE + wave * 16 + (lane & 15)] = (__bf16)x;
      }
  }
  __syncthreads();

  // layer 2: h1 @ W2 (K=128)
  v8f a2[4];
#pragma unroll
  for (int s = 0; s < 4; ++s)
#pragma unroll
    for (int v = 0; v < 8; ++v) a2[s][v] = 0.f;
  for (int kg = 0; kg < 4; ++kg) {
    v16bf bfrag = *(const v16bf*)(p2 + ((size_t)(kg * 8 + wave) * 32 + lane) * 16);
    int koff = kg * 32 + ((lane & 16) ? 8 : 0);
#pragma unroll
    for (int s = 0; s < 4; ++s) {
      const __bf16* ap = h1LDS + (s * 16 + (lane & 15)) * H1STRIDE + koff;
      v8bf lo = *(const v8bf*)ap;
      v8bf hi = *(const v8bf*)(ap + 16);
      v16bf af = __builtin_shufflevector(lo, hi, 0,1,2,3,4,5,6,7,8,9,10,11,12,13,14,15);
      a2[s] = __builtin_amdgcn_wmma_f32_16x16x32_bf16(false, af, false, bfrag,
                                                      (short)0, a2[s], false, false);
    }
  }
  float* h2 = (float*)featLDS;             // reuse 33792B region as 64 x 132 f32
  {
    float bn = b2[wave * 16 + (lane & 15)];
#pragma unroll
    for (int s = 0; s < 4; ++s)
#pragma unroll
      for (int v = 0; v < 8; ++v) {
        int rl = s * 16 + ((lane & 16) ? 8 : 0) + v;
        float x = a2[s][v] + bn;
        x = (x > 0.f) ? x : 0.2f * x;
        h2[rl * H2STRIDE + wave * 16 + (lane & 15)] = x;
      }
  }
  __syncthreads();

  // layer 3 + softmax (tiny)
  if (t < 64) {
    const float* hr = h2 + t * H2STRIDE;
    float l0 = b3[0], l1 = b3[1];
    for (int k = 0; k < 128; ++k) {
      float h = hr[k];
      l0 += h * W3[k * 2];
      l1 += h * W3[k * 2 + 1];
    }
    float m = fmaxf(l0, l1);
    float e0 = __expf(l0 - m), e1 = __expf(l1 - m);
    float s = e0 + e1;
    out[(size_t)(row0 + t) * 2]     = e0 / s;
    out[(size_t)(row0 + t) * 2 + 1] = e1 / s;
  }
}

extern "C" void kernel_launch(void* const* d_in, const int* in_sizes, int n_in,
                              void* d_out, int out_size, void* d_ws, size_t ws_size,
                              hipStream_t stream) {
  (void)in_sizes; (void)n_in; (void)out_size; (void)ws_size;
  const float* A  = (const float*)d_in[0];
  const float* AE = (const float*)d_in[1];
  const float* W1 = (const float*)d_in[2];
  const float* b1 = (const float*)d_in[3];
  const float* W2 = (const float*)d_in[4];
  const float* b2 = (const float*)d_in[5];
  const float* W3 = (const float*)d_in[6];
  const float* b3 = (const float*)d_in[7];
  float* out = (float*)d_out;

  char* ws = (char*)d_ws;
  float*  rstats = (float*)ws;                       // B*4 f32 = 1 MB
  float*  cs_a   = (float*)(ws + 1048576);           // 512 f32
  float*  cs_ae  = cs_a + D_;
  float*  c_a    = cs_ae + D_;
  float*  c_ae   = c_a + D_;
  __bf16* p1     = (__bf16*)(ws + 1048576 + 8192);   // 65*8*32*16 bf16
  __bf16* p2     = p1 + P1ELEMS;                     // 4*8*32*16 bf16

  k_init   <<<2, 256, 0, stream>>>(cs_a, cs_ae);
  k_stats  <<<B_ / 64, 256, 0, stream>>>(A, AE, rstats, cs_a, cs_ae);
  k_centers<<<1, 512, 0, stream>>>(cs_a, cs_ae, c_a, c_ae);
  k_pack   <<<(P1ELEMS + P2ELEMS) / 256, 256, 0, stream>>>(W1, W2, p1, p2);
  k_main   <<<B_ / 64, 256, 0, stream>>>(A, AE, b1, b2, W3, b3, rstats, c_a, c_ae, p1, p2, out);
}
